// TritonGatherConv_14242111553757
// MI455X (gfx1250) — matrix operand order repaired and checked by reading, not compiled, not run on hardware
//
#include <hip/hip_runtime.h>
#include <hip/hip_bf16.h>

#define BB 2
#define LL 1024
#define CC 1024
#define HH 8
#define DD 128            // CC / HH
#define SS 32
#define KK 64

#define PADB 160          // bytes per B row in LDS (128 B data + 32 B pad), 32B aligned

typedef __attribute__((ext_vector_type(16))) _Float16 v16h;
typedef __attribute__((ext_vector_type(8)))  _Float16 v8h;
typedef __attribute__((ext_vector_type(4)))  _Float16 v4h;
typedef __attribute__((ext_vector_type(8)))  float    v8f;

// ---------------------------------------------------------------- convert
__global__ void f32_to_f16_kernel(const float* __restrict__ in,
                                  _Float16* __restrict__ out, int n) {
  int i = blockIdx.x * blockDim.x + threadIdx.x;
  if (i < n) out[i] = (_Float16)in[i];
}

// ---------------------------------------------------------------- wave params (pure fp32 — bin/floor sensitive)
__global__ __launch_bounds__(512) void wave_params_kernel(
    const float* __restrict__ x,       // [B*L, C]
    const float* __restrict__ ww,      // [16, C]
    const float* __restrict__ wb,      // [16]
    float* __restrict__ freq,          // [B*L, H]
    float* __restrict__ phase) {       // [B*L, H]
  int row  = blockIdx.x;
  int wave = threadIdx.x >> 5;         // 0..15
  int lane = threadIdx.x & 31;
  const float* xr = x  + (size_t)row  * CC;
  const float* wr = ww + (size_t)wave * CC;
  float s = 0.f;
  for (int k = lane; k < CC; k += 32) s += xr[k] * wr[k];
  for (int m = 16; m > 0; m >>= 1) s += __shfl_xor(s, m, 32);
  __shared__ float wv[16];
  if (lane == 0) wv[wave] = s + wb[wave];
  __syncthreads();
  if (threadIdx.x < HH) {
    int h = threadIdx.x;
    float f = wv[2 * h], p = wv[2 * h + 1];
    float sig = 1.f / (1.f + expf(-f));
    float fr  = 1.f + sig * 15.f;      // MIN_FREQ + sigmoid*(MAX-MIN)
    float ph  = tanhf(p) * fr;
    freq [row * HH + h] = fr;
    phase[row * HH + h] = ph;
  }
}

// ---------------------------------------------------------------- WMMA GEMM
// C[M,N] (f32) = A[M,K] (f16) @ W[N,K]^T (f16) + bias
// Block: 256 thr = 8 waves -> 128(M) x 64(N) tile. B tile staged in LDS via
// async-to-LDS loads (double buffered), shared by all 8 waves; A fragments
// loaded per-wave straight from global.
__global__ __launch_bounds__(256) void wmma_gemm_f16_kernel(
    const _Float16* __restrict__ A,
    const _Float16* __restrict__ W,
    const float* __restrict__ bias,    // may be nullptr
    float* __restrict__ Cm,
    int M, int N, int Kd) {
  __shared__ __attribute__((aligned(64))) char smemB[2][64 * PADB];

  int nblk = N / 64;
  int bm   = blockIdx.x / nblk;        // 128-row super tile
  int bn   = blockIdx.x % nblk;        // 64-col tile
  int wave = threadIdx.x >> 5;
  int lane = threadIdx.x & 31;
  int hlf  = lane >> 4;                // K sub-block selector
  int idx  = lane & 15;                // M-row (A) / N-col (B,C)
  int t    = threadIdx.x;

  int m0 = bm * 128 + wave * 16;
  int n0 = bn * 64;

  const _Float16* Arow = A + (size_t)(m0 + idx) * Kd;

  v8f z = {0.f, 0.f, 0.f, 0.f, 0.f, 0.f, 0.f, 0.f};
  v8f acc[4];
  for (int q = 0; q < 4; q++) acc[q] = z;

  // Stage B chunk c (k in [c*64, c*64+64)) into smemB[buf] with async loads.
  auto loadB = [&](int c, int buf) {
#pragma unroll
    for (int i = 0; i < 2; i++) {
      int li  = i * 256 + t;                 // 0..511
      int row = li >> 3;                     // 0..63 (local n)
      int seg = li & 7;                      // 16-byte segment within 128B row
      const _Float16* g = W + (size_t)(n0 + row) * Kd + c * 64 + seg * 8;
      unsigned laddr =
          (unsigned)(unsigned long long)&smemB[buf][row * PADB + seg * 16];
      unsigned long long gaddr = (unsigned long long)g;
      asm volatile("global_load_async_to_lds_b128 %0, %1, off"
                   :: "v"(laddr), "v"(gaddr) : "memory");
    }
  };

  auto compute = [&](int c, int buf) {
#pragma unroll
    for (int ks = 0; ks < 2; ks++) {         // two 32-K WMMA steps per chunk
      int k0 = c * 64 + ks * 32;
      // A fragment: a[0..7]=A[m][k0+hlf*16..], a[8..15]=A[m][k0+hlf*16+8..]
      v8h alo = *(const v8h*)(Arow + k0 + hlf * 16);
      v8h ahi = *(const v8h*)(Arow + k0 + hlf * 16 + 8);
      v16h a  = __builtin_shufflevector(alo, ahi,
                  0, 1, 2, 3, 4, 5, 6, 7, 8, 9, 10, 11, 12, 13, 14, 15);
#pragma unroll
      for (int q = 0; q < 4; q++) {
        // B fragment: b[j] = W[n][k0 + hlf*16 + j], contiguous in LDS row
        const char* p =
            &smemB[buf][(q * 16 + idx) * PADB + ks * 64 + hlf * 32];
        v16h bfrag = *(const v16h*)p;
        acc[q] = __builtin_amdgcn_wmma_f32_16x16x32_f16(
            false, a, false, bfrag, (short)0, acc[q], false, false);
      }
    }
  };

  int nchunks = Kd / 64;
  loadB(0, 0);
  asm volatile("s_wait_asynccnt 0x0" ::: "memory");
  __syncthreads();
  for (int c = 0; c < nchunks; c++) {
    int buf = c & 1;
    if (c + 1 < nchunks) loadB(c + 1, buf ^ 1);   // prefetch next chunk
    compute(c, buf);                              // overlaps with async DMA
    if (c + 1 < nchunks)
      asm volatile("s_wait_asynccnt 0x0" ::: "memory");
    __syncthreads();   // all waves done reading buf; next chunk visible
  }

#pragma unroll
  for (int q = 0; q < 4; q++) {
    int n = n0 + q * 16 + idx;
    float bv = bias ? bias[n] : 0.f;
#pragma unroll
    for (int r = 0; r < 8; r++) {
      int m = m0 + hlf * 8 + r;        // C layout: VGPR r -> M = hlf*8 + r
      Cm[(size_t)m * N + n] = acc[q][r] + bv;
    }
  }
}

// ---------------------------------------------------------------- gather + tap-weighted mix
// one wave per (b,l,h): lane = tap s for scalar math, then each lane owns 4 channels
__global__ __launch_bounds__(256) void gather_mix_kernel(
    const float* __restrict__ x,       // [B*L, C] fp32
    const float* __restrict__ freq,    // [B*L, H]
    const float* __restrict__ phase,   // [B*L, H]
    const float* __restrict__ kmat,    // [B*L, H*K] fp32
    _Float16* __restrict__ y) {        // [B*L, C] f16
  int gw   = blockIdx.x * 8 + (threadIdx.x >> 5);
  int lane = threadIdx.x & 31;
  int row  = gw >> 3;                  // b*L + l
  int h    = gw & 7;
  int l    = row & (LL - 1);
  int b    = row >> 10;

  float fr  = freq [row * HH + h];
  float ph  = phase[row * HH + h];
  float off = ph + fr * ((float)lane - 15.5f);   // taps = s - 16 + 0.5
  float pos = (float)l + off;
  pos = fminf(fmaxf(pos, 0.f), (float)(LL - 1));
  float i0f  = floorf(pos);
  float frac = pos - i0f;
  int i0 = (int)i0f;
  int i1 = min(i0 + 1, LL - 1);
  int bin = (int)floorf((off + 256.f) * 0.125f); // (off+MAX_R)/(2*MAX_R)*K
  bin = max(0, min(bin, KK - 1));
  float w = kmat[(size_t)row * (HH * KK) + h * KK + bin];

  int d0 = lane * 4;
  const float* xb = x + (size_t)b * LL * CC + h * DD + d0;
  float a0 = 0.f, a1 = 0.f, a2 = 0.f, a3 = 0.f;
#pragma unroll 4
  for (int s = 0; s < SS; s++) {
    float ws = __shfl(w, s, 32);
    float fs = __shfl(frac, s, 32);
    int   j0 = __shfl(i0, s, 32);
    int   j1 = __shfl(i1, s, 32);
    const float4 g0 = *(const float4*)(xb + (size_t)j0 * CC);
    const float4 g1 = *(const float4*)(xb + (size_t)j1 * CC);
    a0 += ws * (g0.x + fs * (g1.x - g0.x));
    a1 += ws * (g0.y + fs * (g1.y - g0.y));
    a2 += ws * (g0.z + fs * (g1.z - g0.z));
    a3 += ws * (g0.w + fs * (g1.w - g0.w));
  }
  v4h o;
  o[0] = (_Float16)a0; o[1] = (_Float16)a1;
  o[2] = (_Float16)a2; o[3] = (_Float16)a3;
  *(v4h*)(y + (size_t)row * CC + h * DD + d0) = o;
}

// ---------------------------------------------------------------- launch
extern "C" void kernel_launch(void* const* d_in, const int* in_sizes, int n_in,
                              void* d_out, int out_size, void* d_ws, size_t ws_size,
                              hipStream_t stream) {
  const float* x        = (const float*)d_in[0];   // [B,L,C]
  const float* wave_w   = (const float*)d_in[1];   // [16,C]
  const float* wave_b   = (const float*)d_in[2];   // [16]
  const float* kernel_w = (const float*)d_in[3];   // [512,C]
  const float* kernel_b = (const float*)d_in[4];   // [512]
  const float* out_w    = (const float*)d_in[5];   // [C,C]
  float* out = (float*)d_out;                      // [B,L,C]

  const int ROWS = BB * LL;                        // 2048
  char* ws = (char*)d_ws;
  size_t off = 0;
  auto alloc = [&](size_t bytes) -> void* {
    void* p = ws + off;
    off = (off + bytes + 255) & ~(size_t)255;
    return p;
  };
  _Float16* x16   = (_Float16*)alloc((size_t)ROWS * CC * 2);       // 4 MB
  _Float16* kw16  = (_Float16*)alloc((size_t)HH * KK * CC * 2);    // 1 MB
  _Float16* ow16  = (_Float16*)alloc((size_t)CC * CC * 2);         // 2 MB
  float*    freqP = (float*)alloc((size_t)ROWS * HH * 4);
  float*    phasP = (float*)alloc((size_t)ROWS * HH * 4);
  float*    kmat  = (float*)alloc((size_t)ROWS * HH * KK * 4);     // 4 MB
  _Float16* y16   = (_Float16*)alloc((size_t)ROWS * CC * 2);       // 4 MB

  // 1) f32 -> f16 converts
  {
    int n = ROWS * CC;
    f32_to_f16_kernel<<<(n + 255) / 256, 256, 0, stream>>>(x, x16, n);
    n = HH * KK * CC;
    f32_to_f16_kernel<<<(n + 255) / 256, 256, 0, stream>>>(kernel_w, kw16, n);
    n = CC * CC;
    f32_to_f16_kernel<<<(n + 255) / 256, 256, 0, stream>>>(out_w, ow16, n);
  }

  // 2) wave params (fp32): freq/phase per (b,l,h)
  wave_params_kernel<<<ROWS, 512, 0, stream>>>(x, wave_w, wave_b, freqP, phasP);

  // 3) kmat = x @ kernel_w.T + kernel_b   (M=2048, N=512, K=1024)
  {
    int M = ROWS, N = HH * KK;
    int blocks = (M / 128) * (N / 64);             // 128
    wmma_gemm_f16_kernel<<<blocks, 256, 0, stream>>>(x16, kw16, kernel_b,
                                                     kmat, M, N, CC);
  }

  // 4) gather + mix -> y (f16)
  gather_mix_kernel<<<(ROWS * HH) / 8, 256, 0, stream>>>(x, freqP, phasP, kmat, y16);

  // 5) out = y @ out_w.T   (M=2048, N=1024, K=1024), fp32 straight to d_out
  {
    int M = ROWS, N = CC;
    int blocks = (M / 128) * (N / 64);             // 256
    wmma_gemm_f16_kernel<<<blocks, 256, 0, stream>>>(y16, ow16, nullptr,
                                                     out, M, N, CC);
  }
}
